// EquivMPNNLayer_7370163880190
// MI455X (gfx1250) — compile-verified, matrix operand order checked by reading, not run on hardware
//
#include <hip/hip_runtime.h>
#include <hip/hip_bf16.h>
#include <math.h>

// ---------------------------------------------------------------------------
// EquivMPNN layer, fused for MI455X (gfx1250, wave32, WMMA f16->f32).
// B=2, N=256, Cs=128, Cv=64, Er=16; MLP hidden = 256.
// ---------------------------------------------------------------------------

typedef __attribute__((ext_vector_type(16))) _Float16 v16h;
typedef __attribute__((ext_vector_type(8)))  float    v8f;
typedef __attribute__((ext_vector_type(4)))  float    f32x4;   // builtin vec (addrspace-safe)

#define CS   128
#define CV   64
#define ER   16
#define NN   256
#define KP1  288      // 275 padded to multiple of 32
#define HID  256

// Address-space-qualified pointer types so LDS traffic lowers to ds_load_b128
// (32-bit addressing, no generic-pointer null checks) and weight traffic
// lowers to global_load_b128.
typedef __attribute__((address_space(3))) _Float16       lds_f16_t;
typedef __attribute__((address_space(3))) const _Float16 lds_cf16_t;
typedef __attribute__((address_space(3))) const f32x4    lds_cf4_t;
typedef __attribute__((address_space(1))) const _Float16 glb_cf16_t;
typedef __attribute__((address_space(1))) const f32x4    glb_cf4_t;

#if __has_builtin(__builtin_amdgcn_tanhf)
__device__ inline float fast_tanh(float x) { return __builtin_amdgcn_tanhf(x); }
#elif __has_builtin(__builtin_amdgcn_tanh_f32)
__device__ inline float fast_tanh(float x) { return __builtin_amdgcn_tanh_f32(x); }
#else
__device__ inline float fast_tanh(float x) { return tanhf(x); }
#endif

__device__ inline v8f v8f_zero() {
    v8f z = {0.f, 0.f, 0.f, 0.f, 0.f, 0.f, 0.f, 0.f};
    return z;
}

// 16-bit operand fragment, CDNA5 wave32 A-layout (also used for B as W^T):
//   lanes 0-15 : row = row0+lane,     halves 0..7 = K k0+0..7,  8..15 = K k0+16..23
//   lanes 16-31: row = row0+lane-16,  halves 0..7 = K k0+8..15, 8..15 = K k0+24..31
__device__ inline v16h frag_ld_lds(lds_cf16_t* base, int ld, int row0, int k0, int lane) {
    lds_cf16_t* p = base + (row0 + (lane & 15)) * ld + k0 + (lane >> 4) * 8;
    union { v16h v; f32x4 q[2]; } u;
    u.q[0] = *(lds_cf4_t*)p;
    u.q[1] = *(lds_cf4_t*)(p + 16);
    return u.v;
}

__device__ inline v16h frag_ld_glb(glb_cf16_t* base, int ld, int row0, int k0, int lane) {
    glb_cf16_t* p = base + (size_t)(row0 + (lane & 15)) * ld + k0 + (lane >> 4) * 8;
    union { v16h v; f32x4 q[2]; } u;
    u.q[0] = *(glb_cf4_t*)p;
    u.q[1] = *(glb_cf4_t*)(p + 16);
    return u.v;
}

// 64 x (NT*16) tile GEMM accumulate for one wave: A [64 x K] in LDS,
// W^T [N x K] in global.  Per k-step: 4 A-frags + NT B-frags, then 4*NT
// back-to-back WMMAs (independent accumulators -> good overlap).
template <int NT>
struct AccT { v8f a[NT][4]; };

template <int NT>
__device__ inline AccT<NT> gemm_acc(lds_cf16_t* A, int lda, int K,
                                    glb_cf16_t* W, int n0, int lane) {
    AccT<NT> r;
#pragma unroll
    for (int nt = 0; nt < NT; ++nt)
#pragma unroll
        for (int mt = 0; mt < 4; ++mt) r.a[nt][mt] = v8f_zero();

    for (int k0 = 0; k0 < K; k0 += 32) {
        v16h bf[NT];
#pragma unroll
        for (int nt = 0; nt < NT; ++nt)
            bf[nt] = frag_ld_glb(W, K, n0 + nt * 16, k0, lane);
        v16h af[4];
#pragma unroll
        for (int mt = 0; mt < 4; ++mt)
            af[mt] = frag_ld_lds(A, lda, mt * 16, k0, lane);
#pragma unroll
        for (int nt = 0; nt < NT; ++nt)
#pragma unroll
            for (int mt = 0; mt < 4; ++mt)
                r.a[nt][mt] = __builtin_amdgcn_wmma_f32_16x16x32_f16(
                    false, af[mt], false, bf[nt], (short)0, r.a[nt][mt], false, false);
    }
    return r;
}

// Full hidden layer: out[64 x N] = tanh(A @ W^T + bias), N covered by 8 waves * NT.
template <int NT>
__device__ inline void gemm_layer(lds_cf16_t* A, int lda, int K,
                                  glb_cf16_t* W, const float* __restrict__ bias,
                                  lds_f16_t* Out, int ldo, int wave, int lane) {
    int n0 = wave * NT * 16;
    AccT<NT> acc = gemm_acc<NT>(A, lda, K, W, n0, lane);
    int rbase = (lane >> 4) * 8;       // C/D layout: VGPR r -> M = r (+8 hi half)
#pragma unroll
    for (int nt = 0; nt < NT; ++nt) {
        int col = n0 + nt * 16 + (lane & 15);
        float bv = bias[col];
#pragma unroll
        for (int mt = 0; mt < 4; ++mt)
#pragma unroll
            for (int r = 0; r < 8; ++r) {
                float v = fast_tanh(acc.a[nt][mt][r] + bv);
                Out[(mt * 16 + rbase + r) * ldo + col] = (_Float16)v;
            }
    }
}

// ---------------------------------------------------------------------------
// Kernel 0: pack fp32 weight W[K][N] into f16 W^T[N][Kp] (K zero-padded).
// ---------------------------------------------------------------------------
__global__ void pack_w_kernel(const float* __restrict__ W, _Float16* __restrict__ Wt,
                              int K, int N, int Kp) {
    int idx = blockIdx.x * 256 + threadIdx.x;
    if (idx >= N * Kp) return;
    int n = idx / Kp, k = idx - n * Kp;
    Wt[idx] = (k < K) ? (_Float16)W[(size_t)k * N + n] : (_Float16)0.f;
}

// ---------------------------------------------------------------------------
// Kernel 1: one workgroup per (b,i). Builds pair features for all j in LDS,
// runs phi_s and phi_ab chains with WMMA, reduces M_s and M_v over j.
// ---------------------------------------------------------------------------
__global__ __launch_bounds__(256) void pair_kernel(
    const float* __restrict__ h_s, const float* __restrict__ h_v,
    const float* __restrict__ rhat, const float* __restrict__ eij,
    const unsigned char* __restrict__ mask,
    const _Float16* __restrict__ Wt1s, const float* __restrict__ b1s,
    const _Float16* __restrict__ Wt2s, const float* __restrict__ b2s,
    const _Float16* __restrict__ Wt3s, const float* __restrict__ b3s,
    const _Float16* __restrict__ Wt1a, const float* __restrict__ b1a,
    const _Float16* __restrict__ Wt2a, const float* __restrict__ b2a,
    const _Float16* __restrict__ Wt3a, const float* __restrict__ b3a,
    float* __restrict__ Ms_out, float* __restrict__ Mv_out) {

    const int node = blockIdx.x;            // b*N + i
    const int bI   = node >> 8;
    const int tid  = threadIdx.x;
    const int wave = tid >> 5;
    const int lane = tid & 31;

    __shared__ alignas(16) _Float16 sInv[64 * KP1];   // 36.0 KB
    __shared__ alignas(16) _Float16 sH1[64 * HID];    // 32.0 KB
    __shared__ alignas(16) _Float16 sH2[64 * HID];    // 32.0 KB
    __shared__ float sHi[CS];
    __shared__ float sHvi[3 * CV];
    __shared__ float sSumI[3];
    __shared__ float sMask[64];
    __shared__ float sMs[CS];
    __shared__ float sMv[3 * CV];

    lds_cf16_t* cInv = (lds_cf16_t*)sInv;
    lds_cf16_t* cH1  = (lds_cf16_t*)sH1;
    lds_cf16_t* cH2  = (lds_cf16_t*)sH2;
    glb_cf16_t* gW1s = (glb_cf16_t*)Wt1s;
    glb_cf16_t* gW2s = (glb_cf16_t*)Wt2s;
    glb_cf16_t* gW3s = (glb_cf16_t*)Wt3s;
    glb_cf16_t* gW1a = (glb_cf16_t*)Wt1a;
    glb_cf16_t* gW2a = (glb_cf16_t*)Wt2a;
    glb_cf16_t* gW3a = (glb_cf16_t*)Wt3a;

    if (tid < CS)  { sHi[tid]  = h_s[(size_t)node * CS + tid];  sMs[tid] = 0.f; }
    if (tid < 192) { sHvi[tid] = h_v[(size_t)node * 192 + tid]; sMv[tid] = 0.f; }
    __syncthreads();
    if (tid < 3) {
        float s = 0.f;
        for (int c = 0; c < CV; ++c) s += sHvi[tid * CV + c];
        sSumI[tid] = s;
    }
    __syncthreads();

    for (int chunk = 0; chunk < 4; ++chunk) {
        const int jbase = chunk * 64;

        // ---- build inv[64 x 288] in LDS (f16) + mask chunk -----------------
        {
            int row = tid >> 2, part = tid & 3;
            int j = jbase + row;
            lds_f16_t* dst = (lds_f16_t*)sInv + row * KP1;
            const float* hsj = h_s + ((size_t)bI * NN + j) * CS;
#pragma unroll 4
            for (int t = 0; t < 32; ++t)
                dst[part * 32 + t] = (_Float16)sHi[part * 32 + t];         // hi
#pragma unroll 4
            for (int t = 0; t < 32; ++t)
                dst[CS + part * 32 + t] = (_Float16)hsj[part * 32 + t];    // hj
            if (part == 0) {
                const float* ep = eij + ((size_t)node * NN + j) * ER;
                for (int t = 0; t < ER; ++t) dst[256 + t] = (_Float16)ep[t];
            } else if (part == 1) {
                for (int k = 275; k < KP1; ++k) dst[k] = (_Float16)0.f;    // pad
            } else if (part == 2) {
                sMask[row] = mask[(size_t)node * NN + j] ? 1.f : 0.f;
            } else {
                const float* hvj = h_v + ((size_t)bI * NN + j) * 192;
                float c1 = 0.f, s0 = 0.f, s1 = 0.f, s2 = 0.f;
                for (int t = 0; t < 64; ++t)   { float v = hvj[t]; c1 += sHvi[t] * v; s0 += v; }
                for (int t = 64; t < 128; ++t) { float v = hvj[t]; c1 += sHvi[t] * v; s1 += v; }
                for (int t = 128; t < 192; ++t){ float v = hvj[t]; c1 += sHvi[t] * v; s2 += v; }
                const float* rp = rhat + ((size_t)node * NN + j) * 3;
                float r0 = rp[0], r1 = rp[1], r2 = rp[2];
                dst[272] = (_Float16)c1;
                dst[273] = (_Float16)(sSumI[0] * r0 + sSumI[1] * r1 + sSumI[2] * r2);
                dst[274] = (_Float16)(s0 * r0 + s1 * r1 + s2 * r2);
            }
        }
        __syncthreads();

        // ---- chain phi_s: 288 -> 256(tanh) -> 256(tanh) -> 128 -------------
        gemm_layer<2>(cInv, KP1, KP1, gW1s, b1s, (lds_f16_t*)sH1, HID, wave, lane);
        __syncthreads();
        gemm_layer<2>(cH1, HID, HID, gW2s, b2s, (lds_f16_t*)sH2, HID, wave, lane);
        __syncthreads();
        {   // layer 3 (no tanh) + mask + sum over j -> sMs
            int n0 = wave * 16;                          // 8 waves x 16 = 128 cols
            AccT<1> acc = gemm_acc<1>(cH2, HID, HID, gW3s, n0, lane);
            int col = n0 + (lane & 15);
            int rbase = (lane >> 4) * 8;
            float bv = b3s[col];
            float partial = 0.f;
#pragma unroll
            for (int mt = 0; mt < 4; ++mt)
#pragma unroll
                for (int r = 0; r < 8; ++r) {
                    int jl = mt * 16 + rbase + r;
                    partial += (acc.a[0][mt][r] + bv) * sMask[jl];
                }
            partial += __shfl_xor(partial, 16, 32);      // combine lane L / L+16
            if (lane < 16) sMs[col] += partial;          // unique writer per col
        }
        __syncthreads();

        // ---- chain phi_ab: 288 -> 256(tanh) -> 256(tanh) -> 128 ------------
        gemm_layer<2>(cInv, KP1, KP1, gW1a, b1a, (lds_f16_t*)sH1, HID, wave, lane);
        __syncthreads();
        gemm_layer<2>(cH1, HID, HID, gW2a, b2a, (lds_f16_t*)sH2, HID, wave, lane);
        __syncthreads();
        {   // layer 3: alpha/beta -> M_v contraction (channel-wise, LDS atomics)
            int n0 = wave * 16;
            AccT<1> acc = gemm_acc<1>(cH2, HID, HID, gW3a, n0, lane);
            int col = n0 + (lane & 15);                  // 0..127
            int rbase = (lane >> 4) * 8;
            float bv = b3a[col];
            if (col < CV) {                              // alpha * h_v[b,j,d,c]
#pragma unroll
                for (int mt = 0; mt < 4; ++mt)
#pragma unroll
                    for (int r = 0; r < 8; ++r) {
                        int jl = mt * 16 + rbase + r;
                        int j  = jbase + jl;
                        float a = (acc.a[0][mt][r] + bv) * sMask[jl];
                        const float* hvj = h_v + ((size_t)bI * NN + j) * 192 + col;
                        atomicAdd(&sMv[col],          a * hvj[0]);
                        atomicAdd(&sMv[CV + col],     a * hvj[CV]);
                        atomicAdd(&sMv[2 * CV + col], a * hvj[2 * CV]);
                    }
            } else {                                     // beta * rhat[b,i,j,d]
                int c = col - CV;
#pragma unroll
                for (int mt = 0; mt < 4; ++mt)
#pragma unroll
                    for (int r = 0; r < 8; ++r) {
                        int jl = mt * 16 + rbase + r;
                        int j  = jbase + jl;
                        float bt = (acc.a[0][mt][r] + bv) * sMask[jl];
                        const float* rp = rhat + ((size_t)node * NN + j) * 3;
                        atomicAdd(&sMv[c],           bt * rp[0]);
                        atomicAdd(&sMv[CV + c],      bt * rp[1]);
                        atomicAdd(&sMv[2 * CV + c],  bt * rp[2]);
                    }
            }
        }
        __syncthreads();
    }

    if (tid < CS)  Ms_out[(size_t)node * CS + tid]  = sMs[tid];
    if (tid < 192) Mv_out[(size_t)node * 192 + tid] = sMv[tid];
}

// ---------------------------------------------------------------------------
// Kernel 2: per-node update MLPs (tiny; VALU). One workgroup per node.
// ---------------------------------------------------------------------------
__global__ __launch_bounds__(256) void update_kernel(
    const float* __restrict__ h_s, const float* __restrict__ h_v,
    const float* __restrict__ Ms, const float* __restrict__ Mv,
    const float* __restrict__ Wu0, const float* __restrict__ bu0,
    const float* __restrict__ Wu1, const float* __restrict__ bu1,
    const float* __restrict__ Wg0, const float* __restrict__ bg0,
    const float* __restrict__ Wg1, const float* __restrict__ bg1,
    float* __restrict__ out) {

    const int node = blockIdx.x;
    const int tid  = threadIdx.x;

    __shared__ float upd[2 * CS + 3];    // [h_s | M_s | nv | nMv | dot]
    __shared__ float hid[HID];
    __shared__ float ghid[HID];
    __shared__ float red[3];
    __shared__ float gsh[CV];

    if (tid < CS)       upd[tid] = h_s[(size_t)node * CS + tid];
    else if (tid < 256) upd[tid] = Ms[(size_t)node * CS + (tid - CS)];
    if (tid < 3) red[tid] = 0.f;
    __syncthreads();
    if (tid < 192) {
        float v = h_v[(size_t)node * 192 + tid];
        float m = Mv[(size_t)node * 192 + tid];
        atomicAdd(&red[0], v * v);
        atomicAdd(&red[1], m * m);
        atomicAdd(&red[2], v * m);
    }
    __syncthreads();
    if (tid == 0) {
        upd[2 * CS]     = sqrtf(fmaxf(red[0], 1e-12f));
        upd[2 * CS + 1] = sqrtf(fmaxf(red[1], 1e-12f));
        upd[2 * CS + 2] = red[2];
    }
    __syncthreads();

    {   // hidden layers of U_s and gate (259 -> 256, tanh)
        float a = bu0[tid], g = bg0[tid];
        for (int k = 0; k < 2 * CS + 3; ++k) {
            float u = upd[k];
            a += u * Wu0[(size_t)k * 256 + tid];
            g += u * Wg0[(size_t)k * 256 + tid];
        }
        hid[tid]  = fast_tanh(a);
        ghid[tid] = fast_tanh(g);
    }
    __syncthreads();

    if (tid < CS) {     // dh + residual -> h_s_new
        float a = bu1[tid];
        for (int h = 0; h < HID; ++h) a += hid[h] * Wu1[(size_t)h * CS + tid];
        out[(size_t)node * CS + tid] = upd[tid] + a;
    }
    if (tid < CV) {     // g = tanh(gate mlp)
        float g = bg1[tid];
        for (int h = 0; h < HID; ++h) g += ghid[h] * Wg1[(size_t)h * CV + tid];
        gsh[tid] = fast_tanh(g);
    }
    __syncthreads();

    if (tid < 192) {    // h_v_new = h_v + g[c] * M_v
        size_t idx = (size_t)node * 192 + tid;
        out[(size_t)2 * NN * CS + idx] = h_v[idx] + gsh[tid & (CV - 1)] * Mv[idx];
    }
}

// ---------------------------------------------------------------------------
extern "C" void kernel_launch(void* const* d_in, const int* in_sizes, int n_in,
                              void* d_out, int out_size, void* d_ws, size_t ws_size,
                              hipStream_t stream) {
    (void)in_sizes; (void)n_in; (void)out_size; (void)ws_size;

    const float* h_s  = (const float*)d_in[0];
    const float* h_v  = (const float*)d_in[1];
    const float* rhat = (const float*)d_in[2];
    const float* eij  = (const float*)d_in[3];
    const unsigned char* pmask = (const unsigned char*)d_in[4];   // bool array
    // phi_s: W0(275,256) b0 W1(256,256) b1 W2(256,128) b2
    const float* Ws0 = (const float*)d_in[5];
    const float* bs0 = (const float*)d_in[6];
    const float* Ws1 = (const float*)d_in[7];
    const float* bs1 = (const float*)d_in[8];
    const float* Ws2 = (const float*)d_in[9];
    const float* bs2 = (const float*)d_in[10];
    // phi_ab
    const float* Wa0 = (const float*)d_in[11];
    const float* ba0 = (const float*)d_in[12];
    const float* Wa1 = (const float*)d_in[13];
    const float* ba1 = (const float*)d_in[14];
    const float* Wa2 = (const float*)d_in[15];
    const float* ba2 = (const float*)d_in[16];
    // U_s: W0(259,256) b0 W1(256,128) b1
    const float* Wu0 = (const float*)d_in[17];
    const float* bu0 = (const float*)d_in[18];
    const float* Wu1 = (const float*)d_in[19];
    const float* bu1 = (const float*)d_in[20];
    // gate: W0(259,256) b0 W1(256,64) b1
    const float* Wg0 = (const float*)d_in[21];
    const float* bg0 = (const float*)d_in[22];
    const float* Wg1 = (const float*)d_in[23];
    const float* bg1 = (const float*)d_in[24];

    char* ws = (char*)d_ws;
    _Float16* Wt1s = (_Float16*)(ws + 0);        // 256*288 f16 = 147456 B
    _Float16* Wt2s = (_Float16*)(ws + 147456);   // 256*256 f16 = 131072 B
    _Float16* Wt3s = (_Float16*)(ws + 278528);   // 128*256 f16 =  65536 B
    _Float16* Wt1a = (_Float16*)(ws + 344064);
    _Float16* Wt2a = (_Float16*)(ws + 491520);
    _Float16* Wt3a = (_Float16*)(ws + 622592);
    float*    Ms_ws = (float*)(ws + 688128);     // 512*128 f32 = 262144 B
    float*    Mv_ws = (float*)(ws + 950272);     // 512*192 f32 = 393216 B

    // pack weights: W[K][N] fp32 -> W^T[N][Kp] f16, zero-padded K
    pack_w_kernel<<<288, 256, 0, stream>>>(Ws0, Wt1s, 275, 256, KP1);
    pack_w_kernel<<<256, 256, 0, stream>>>(Ws1, Wt2s, 256, 256, 256);
    pack_w_kernel<<<128, 256, 0, stream>>>(Ws2, Wt3s, 256, 128, 256);
    pack_w_kernel<<<288, 256, 0, stream>>>(Wa0, Wt1a, 275, 256, KP1);
    pack_w_kernel<<<256, 256, 0, stream>>>(Wa1, Wt2a, 256, 256, 256);
    pack_w_kernel<<<128, 256, 0, stream>>>(Wa2, Wt3a, 256, 128, 256);

    // fused pair-MLP + reductions: one workgroup per (b,i)
    pair_kernel<<<2 * NN, 256, 0, stream>>>(
        h_s, h_v, rhat, eij, pmask,
        Wt1s, bs0, Wt2s, bs1, Wt3s, bs2,
        Wt1a, ba0, Wt2a, ba1, Wt3a, ba2,
        Ms_ws, Mv_ws);

    // per-node update
    update_kernel<<<2 * NN, 256, 0, stream>>>(
        h_s, h_v, Ms_ws, Mv_ws,
        Wu0, bu0, Wu1, bu1, Wg0, bg0, Wg1, bg1,
        (float*)d_out);
}
